// HardMoE_47802986004697
// MI455X (gfx1250) — compile-verified, matrix operand order checked by reading, not run on hardware
//
#include <hip/hip_runtime.h>

// ---------------- problem constants (B=4, S=2048, D=4096, H=4096, E=8, K=2) --
#define NTOK   8192      // B*S tokens
#define DDIM   4096
#define HDIM   4096
#define NEXP   8
#define MAXPE  NTOK      // worst-case tokens per expert

#define MT     32        // tokens per block  (2 WMMA M-subtiles)
#define NTC    512       // H columns per block (8 waves x 4 N-subtiles of 16)
#define KC     32        // K chunk per WMMA (bf16 16x16x32)
#define NKC    (DDIM / KC)        // 128 K-chunks
#define NSUB   (HDIM / 16)        // 256 N-subtiles per expert
#define PFD    4                  // prefetch distance (K-chunks)

typedef __attribute__((ext_vector_type(16))) __bf16 v16bf;
typedef __attribute__((ext_vector_type(8)))  __bf16 v8bf;
typedef __attribute__((ext_vector_type(8)))  float  v8f;

union V16U { v16bf v; v8bf h[2]; };

// ---------------------------------------------------------------- zero kernel
__global__ void zero_kernel(float* __restrict__ out, int n, int* __restrict__ counts) {
  int i = blockIdx.x * blockDim.x + threadIdx.x;
  if (i < NEXP) counts[i] = 0;
  float4* o4 = (float4*)out;
  int n4 = n >> 2;
  float4 z; z.x = 0.f; z.y = 0.f; z.z = 0.f; z.w = 0.f;
  for (int j = i; j < n4; j += gridDim.x * blockDim.x) o4[j] = z;
}

// --------------------------------------------------- pack x: fp32 -> bf16 rows
__global__ __launch_bounds__(256)
void pack_x_kernel(const float* __restrict__ x, __bf16* __restrict__ xb) {
  int i = (blockIdx.x * blockDim.x + threadIdx.x) * 8;
  float4 a = *(const float4*)(x + i);
  float4 b = *(const float4*)(x + i + 4);
  __bf16 r[8];
  r[0] = (__bf16)a.x; r[1] = (__bf16)a.y; r[2] = (__bf16)a.z; r[3] = (__bf16)a.w;
  r[4] = (__bf16)b.x; r[5] = (__bf16)b.y; r[6] = (__bf16)b.z; r[7] = (__bf16)b.w;
  *(v8bf*)(xb + i) = *(v8bf*)r;
}

// ---------------- pack W: fp32 [E][K][N] -> bf16 fragments [E][nsub][kc][lane][16]
// Fragment element i of lane l maps to K = kc*32 + (((i&8)<<1)|(i&7)) + ((l>>4)<<3),
// N = nsub*16 + (l&15)   (ISA 7.12.2 16-bit B layout, lane = N column).
__global__ __launch_bounds__(256)
void pack_w_kernel(const float* __restrict__ We, __bf16* __restrict__ wb) {
  int id = blockIdx.x * blockDim.x + threadIdx.x;   // one fragment row (32B out)
  int lane = id & 31;
  int t    = id >> 5;
  int kc   = t & (NKC - 1);
  t >>= 7;                                          // NKC = 128
  int nsub = t & (NSUB - 1);
  int e    = t >> 8;                                // NSUB = 256
  int khi  = (lane >> 4) << 3;
  const float* src = We + (size_t)e * DDIM * HDIM
                        + (size_t)(kc * KC + khi) * HDIM
                        + nsub * 16 + (lane & 15);
  float v0[8], v1[8];
#pragma unroll
  for (int j = 0; j < 8; ++j) v0[j] = src[(size_t)j * HDIM];
#pragma unroll
  for (int j = 0; j < 8; ++j) v1[j] = src[(size_t)(16 + j) * HDIM];
  __bf16 r[16];
#pragma unroll
  for (int j = 0; j < 8; ++j) { r[j] = (__bf16)v0[j]; r[8 + j] = (__bf16)v1[j]; }
  __bf16* dst = wb + (size_t)id * 16;
  *(v8bf*)dst       = *(v8bf*)r;
  *(v8bf*)(dst + 8) = *(v8bf*)(r + 8);
}

// ------------------------------------------------- gate + top-2 scatter kernel
__global__ __launch_bounds__(256)
void gate_scatter_kernel(const float* __restrict__ x,
                         const float* __restrict__ Wg,   // [D, E]
                         const float* __restrict__ bg,   // [E]
                         int* __restrict__ counts,       // [E]
                         int* __restrict__ lists) {      // [E][MAXPE]
  const int tok = blockIdx.x;
  const int tid = threadIdx.x;
  float acc[NEXP];
#pragma unroll
  for (int e = 0; e < NEXP; ++e) acc[e] = 0.f;
  const float* xr = x + (size_t)tok * DDIM;
  for (int d = tid; d < DDIM; d += 256) {
    float xv = xr[d];
#pragma unroll
    for (int e = 0; e < NEXP; ++e) acc[e] += xv * Wg[d * NEXP + e];
  }
  __shared__ float red[NEXP * 256];
#pragma unroll
  for (int e = 0; e < NEXP; ++e) red[e * 256 + tid] = acc[e];
  __syncthreads();
  for (int s = 128; s > 0; s >>= 1) {
    if (tid < s) {
#pragma unroll
      for (int e = 0; e < NEXP; ++e) red[e * 256 + tid] += red[e * 256 + tid + s];
    }
    __syncthreads();
  }
  if (tid == 0) {
    float lg[NEXP];
#pragma unroll
    for (int e = 0; e < NEXP; ++e) lg[e] = red[e * 256] + bg[e];
    int b1 = 0;
#pragma unroll
    for (int e = 1; e < NEXP; ++e) if (lg[e] > lg[b1]) b1 = e;   // ties -> lowest idx
    int b2 = -1;
#pragma unroll
    for (int e = 0; e < NEXP; ++e)
      if (e != b1 && (b2 < 0 || lg[e] > lg[b2])) b2 = e;
    int p1 = atomicAdd(&counts[b1], 1);
    lists[b1 * MAXPE + p1] = tok;
    int p2 = atomicAdd(&counts[b2], 1);
    lists[b2 * MAXPE + p2] = tok;
  }
}

// ------------------------------------------------------- main WMMA MoE kernel
// No LDS staging, no in-loop barriers: A fragments load straight from packed
// bf16 x rows, B fragments straight from pre-swizzled bf16 weights.
// Per wave: 2 M-subtiles x 4 N-subtiles -> 8 WMMAs per K-chunk, 12 b128 loads.
__global__ __launch_bounds__(256)
void moe_gemm_kernel(const __bf16* __restrict__ xb,     // [NTOK, D] bf16
                     const __bf16* __restrict__ wb,     // packed fragments
                     const float*  __restrict__ be,     // [E, H]
                     const int*    __restrict__ counts, // [E]
                     const int*    __restrict__ lists,  // [E][MAXPE]
                     float* __restrict__ out) {         // [NTOK, H]
  const int e   = blockIdx.z;
  const int cnt = counts[e];
  const int m0  = blockIdx.x * MT;
  if (m0 >= cnt) return;                 // block-uniform early exit
  const int n0  = blockIdx.y * NTC;

  const int tid  = threadIdx.x;
  const int lane = tid & 31;
  const int wave = tid >> 5;             // 0..7, each owns 64 output columns

  __shared__ int s_tok[MT];
  if (tid < MT) {
    int idx = m0 + tid;
    s_tok[tid] = (idx < cnt) ? lists[e * MAXPE + idx] : -1;
  }
  __syncthreads();

  // Per-lane A row pointers (ISA A layout: lane l -> row M=l&15, khi by lane half).
  const int khi  = (lane >> 4) << 3;
  const int tA0  = s_tok[lane & 15];
  const int tA1  = s_tok[16 + (lane & 15)];
  const __bf16* pa0 = xb + (size_t)(tA0 < 0 ? 0 : tA0) * DDIM + khi;
  const __bf16* pa1 = xb + (size_t)(tA1 < 0 ? 0 : tA1) * DDIM + khi;

  // B fragment tiles: wb[((e*NSUB + nsub)*NKC + kc)*512 + lane*16]
  const int nsub0 = (n0 >> 4) + wave * 4;
  const __bf16* pb0 = wb + ((size_t)(e * NSUB + nsub0) * NKC) * 512 + lane * 16;
  const __bf16* pb1 = pb0 + 1 * (size_t)NKC * 512;
  const __bf16* pb2 = pb0 + 2 * (size_t)NKC * 512;
  const __bf16* pb3 = pb0 + 3 * (size_t)NKC * 512;

  v8f cz = {0.f,0.f,0.f,0.f,0.f,0.f,0.f,0.f};
  v8f c00 = cz, c01 = cz, c02 = cz, c03 = cz;
  v8f c10 = cz, c11 = cz, c12 = cz, c13 = cz;

  auto body = [&](int kc, bool pf) {
    const int k0 = kc * KC;
    V16U a0, a1;
    a0.h[0] = *(const v8bf*)(pa0 + k0);        // K = khi..khi+7
    a0.h[1] = *(const v8bf*)(pa0 + k0 + 16);   // K = khi+16..khi+23
    a1.h[0] = *(const v8bf*)(pa1 + k0);
    a1.h[1] = *(const v8bf*)(pa1 + k0 + 16);
    v16bf b0 = *(const v16bf*)(pb0 + (size_t)kc * 512);
    v16bf b1 = *(const v16bf*)(pb1 + (size_t)kc * 512);
    v16bf b2 = *(const v16bf*)(pb2 + (size_t)kc * 512);
    v16bf b3 = *(const v16bf*)(pb3 + (size_t)kc * 512);
    if (pf) {
      __builtin_prefetch(pb0 + (size_t)(kc + PFD) * 512, 0, 0);
      __builtin_prefetch(pb1 + (size_t)(kc + PFD) * 512, 0, 0);
      __builtin_prefetch(pb2 + (size_t)(kc + PFD) * 512, 0, 0);
      __builtin_prefetch(pb3 + (size_t)(kc + PFD) * 512, 0, 0);
    }
    c00 = __builtin_amdgcn_wmma_f32_16x16x32_bf16(false, a0.v, false, b0,
                                                  (short)0, c00, false, false);
    c01 = __builtin_amdgcn_wmma_f32_16x16x32_bf16(false, a0.v, false, b1,
                                                  (short)0, c01, false, false);
    c02 = __builtin_amdgcn_wmma_f32_16x16x32_bf16(false, a0.v, false, b2,
                                                  (short)0, c02, false, false);
    c03 = __builtin_amdgcn_wmma_f32_16x16x32_bf16(false, a0.v, false, b3,
                                                  (short)0, c03, false, false);
    c10 = __builtin_amdgcn_wmma_f32_16x16x32_bf16(false, a1.v, false, b0,
                                                  (short)0, c10, false, false);
    c11 = __builtin_amdgcn_wmma_f32_16x16x32_bf16(false, a1.v, false, b1,
                                                  (short)0, c11, false, false);
    c12 = __builtin_amdgcn_wmma_f32_16x16x32_bf16(false, a1.v, false, b2,
                                                  (short)0, c12, false, false);
    c13 = __builtin_amdgcn_wmma_f32_16x16x32_bf16(false, a1.v, false, b3,
                                                  (short)0, c13, false, false);
  };

  for (int kc = 0; kc < NKC - PFD; ++kc) body(kc, true);   // branch-free prefetch
#pragma unroll
  for (int kc = NKC - PFD; kc < NKC; ++kc) body(kc, false);

  // ---- epilogue: bias + ReLU + 0.5 (mean of top-2), atomic add into out.
  // C/D layout: VGPR v, lane l -> M = v + 8*(l>>4), N = l&15.
  const float* brow = be + (size_t)e * HDIM;
  v8f acc[2][4] = {{c00, c01, c02, c03}, {c10, c11, c12, c13}};
#pragma unroll
  for (int mt = 0; mt < 2; ++mt) {
#pragma unroll
    for (int ns = 0; ns < 4; ++ns) {
      int col  = n0 + wave * 64 + ns * 16 + (lane & 15);
      float bias = brow[col];
#pragma unroll
      for (int v = 0; v < 8; ++v) {
        int m   = v + 8 * (lane >> 4) + mt * 16;
        int tok = s_tok[m];
        if (tok >= 0) {
          float val = acc[mt][ns][v] + bias;
          val = (val > 0.f) ? val * 0.5f : 0.f;
          unsafeAtomicAdd(&out[(size_t)tok * HDIM + col], val);
        }
      }
    }
  }
}

// ----------------------------------------------------------------------------
extern "C" void kernel_launch(void* const* d_in, const int* in_sizes, int n_in,
                              void* d_out, int out_size, void* d_ws, size_t ws_size,
                              hipStream_t stream) {
  const float* x  = (const float*)d_in[0];   // [4,2048,4096]
  const float* Wg = (const float*)d_in[1];   // [4096,8]
  const float* bg = (const float*)d_in[2];   // [8]
  const float* We = (const float*)d_in[3];   // [8,4096,4096]
  const float* be = (const float*)d_in[4];   // [8,4096]
  float* out = (float*)d_out;                // [4,2048,4096]

  // ---- workspace layout (bytes) ----
  char* ws = (char*)d_ws;
  int* counts = (int*)ws;                                  // 64 B
  int* lists  = (int*)(ws + 64);                           // 8*8192*4 = 256 KB
  size_t off  = 64 + (size_t)NEXP * MAXPE * 4;
  off = (off + 255) & ~(size_t)255;
  __bf16* xb = (__bf16*)(ws + off);                        // 64 MB
  off += (size_t)NTOK * DDIM * 2;
  __bf16* wb = (__bf16*)(ws + off);                        // 256 MB

  zero_kernel<<<2048, 256, 0, stream>>>(out, out_size, counts);
  pack_x_kernel<<<(NTOK * DDIM / 8) / 256, 256, 0, stream>>>(x, xb);
  pack_w_kernel<<<(NEXP * NSUB * NKC * 32) / 256, 256, 0, stream>>>(We, wb);
  gate_scatter_kernel<<<NTOK, 256, 0, stream>>>(x, Wg, bg, counts, lists);

  dim3 grid(NTOK / MT, HDIM / NTC, NEXP);    // m_tile fastest -> weight-slab L2 reuse
  moe_gemm_kernel<<<grid, 256, 0, stream>>>(xb, wb, be, counts, lists, out);
}
